// BalancedAtchleyAttention_11201274708463
// MI455X (gfx1250) — matrix operand order repaired and checked by reading, not compiled, not run on hardware
//
#include <hip/hip_runtime.h>

// ---------------------------------------------------------------------------
// Types for CDNA5 WMMA (wave32): D(f32 16x16) = A(bf16 16x32) x B(bf16 32x16) + C
// ---------------------------------------------------------------------------
typedef __attribute__((ext_vector_type(16))) __bf16 bf16x16;
typedef __attribute__((ext_vector_type(8)))  __bf16 bf16x8;
typedef __attribute__((ext_vector_type(8)))  float  f32x8;
typedef __attribute__((ext_vector_type(4)))  unsigned u32x4;
typedef __attribute__((ext_vector_type(8)))  int      i32x8;
typedef __attribute__((ext_vector_type(4)))  int      i32x4;

union FragU {
    bf16x16 f;
    bf16x8  h[2];
};

__device__ __forceinline__ f32x8 wmma_bf16(bf16x16 a, bf16x16 b, f32x8 c) {
    return __builtin_amdgcn_wmma_f32_16x16x32_bf16(
        false, a, false, b, (short)0, c, false, false);
}

// f32 -> bf16 round-to-nearest-even (finite inputs)
__device__ __forceinline__ unsigned short f2bf(float x) {
    union { float f; unsigned u; } v; v.f = x;
    unsigned u = v.u;
    u += 0x7FFFu + ((u >> 16) & 1u);
    return (unsigned short)(u >> 16);
}
__device__ __forceinline__ unsigned pk2(float a, float b) {
    return (unsigned)f2bf(a) | ((unsigned)f2bf(b) << 16);
}

// ---------------------------------------------------------------------------
// CDNA5 async global->LDS DMA (ASYNCcnt) + TDM tensor DMA (TENSORcnt)
// ---------------------------------------------------------------------------
__device__ __forceinline__ unsigned lds_off(const void* p) {
    return (unsigned)(unsigned long long)(p);  // low 32 bits = LDS byte offset
}
__device__ __forceinline__ void async_g2l_b128(unsigned lds_addr, const void* sbase,
                                               unsigned byte_off) {
    asm volatile("global_load_async_to_lds_b128 %0, %1, %2"
                 :
                 : "v"(lds_addr), "v"(byte_off), "s"(sbase)
                 : "memory");
}
__device__ __forceinline__ void wait_async0(void) {
#if __has_builtin(__builtin_amdgcn_s_wait_asynccnt)
    __builtin_amdgcn_s_wait_asynccnt(0);
#else
    asm volatile("s_wait_asynccnt 0x0" ::: "memory");
#endif
}
__device__ __forceinline__ void wait_tensor0(void) {
#if __has_builtin(__builtin_amdgcn_s_wait_tensorcnt)
    __builtin_amdgcn_s_wait_tensorcnt(0);
#else
    asm volatile("s_wait_tensorcnt 0x0" ::: "memory");
#endif
}

#if __has_builtin(__builtin_amdgcn_tensor_load_to_lds)
#define HAS_TDM 1
#else
#define HAS_TDM 0
#endif

// TDM load of one K chunk: 2D tile, 8B elements, 16 units (=128B) per row,
// 32 rows, with LDS padding of 4 dwords (16B) every 32 dwords (128B) ->
// exactly the STR64 padded layout. Issued by one wave only.
__device__ __forceinline__ void tdm_load_k(unsigned lds_byte, const void* gptr) {
#if HAS_TDM
    unsigned long long ga = (unsigned long long)gptr;
    u32x4 g0;
    g0[0] = 1u;                                             // count=1, user D#
    g0[1] = lds_byte;                                       // lds_addr
    g0[2] = (unsigned)ga;                                   // global_addr[31:0]
    g0[3] = (unsigned)((ga >> 32) & 0x01FFFFFFu) | 0x80000000u; // addr[56:32]|type=2
    i32x8 g1;
    g1[0] = (int)((3u << 16) | (1u << 20) | (4u << 22) | (3u << 25));
            // data_size=8B, pad_enable, pad_interval=32dw, pad_amount=4dw
    g1[1] = (int)(16u << 16);   // tensor_dim0 = 16 units (low half)
    g1[2] = (int)(32u << 16);   // tensor_dim0 hi=0 ; tensor_dim1 = 32 (low half)
    g1[3] = (int)(16u << 16);   // tensor_dim1 hi=0 ; tile_dim0 = 16
    g1[4] = (int)(32u);         // tile_dim1 = 32 ; tile_dim2 = 0
    g1[5] = (int)(16u);         // tensor_dim0_stride = 16 units
    g1[6] = 0;
    g1[7] = 0;
    i32x4 zz = {0, 0, 0, 0};
#if __clang_major__ >= 23
    i32x8 z8 = {0, 0, 0, 0, 0, 0, 0, 0};
    __builtin_amdgcn_tensor_load_to_lds(g0, g1, zz, zz, z8, 0);
#else
    __builtin_amdgcn_tensor_load_to_lds(g0, g1, zz, zz, 0);
#endif
#endif
}

#define DIMC 1024
#define HN   16
#define HD   64
#define LSEQ 512

// LDS row strides (halfwords): 16 distinct banks per half-wave
#define STR32 40   // 32 payload + 8 pad (80B rows)
#define STR64 72   // 64 payload + 8 pad (144B rows)

// workspace byte offsets (total 58MB)
#define QB_OFF  (0u)
#define KB_OFF  (8u  * 1024u * 1024u)
#define VT_OFF  (16u * 1024u * 1024u)
#define OB_OFF  (24u * 1024u * 1024u)
#define C1_OFF  (32u * 1024u * 1024u)   // 1.25MB
#define XB1_OFF (34u * 1024u * 1024u)
#define XB2_OFF (42u * 1024u * 1024u)
#define WQ_OFF  (50u * 1024u * 1024u)
#define WK_OFF  (52u * 1024u * 1024u)
#define WV_OFF  (54u * 1024u * 1024u)
#define WO_OFF  (56u * 1024u * 1024u)

// ---------------------------------------------------------------------------
// One-time f32 -> bf16 conversion (bandwidth-bound)
// ---------------------------------------------------------------------------
__global__ __launch_bounds__(256) void cvt_kernel(const float* __restrict__ src,
                                                  unsigned short* __restrict__ dst) {
    int idx = blockIdx.x * 256 + threadIdx.x;
    const float4* s = (const float4*)src + (size_t)idx * 2;
    float4 a = s[0], b = s[1];
    uint4 u = make_uint4(pk2(a.x, a.y), pk2(a.z, a.w), pk2(b.x, b.y), pk2(b.z, b.w));
    ((uint4*)dst)[idx] = u;
}

// ---------------------------------------------------------------------------
// c1[b,h,i,q] = sum_p atc1[b,i,p] * U[h,p,q]
// ---------------------------------------------------------------------------
__global__ __launch_bounds__(128) void c1_kernel(const float* __restrict__ atc1,
                                                 const float* __restrict__ U,
                                                 float* __restrict__ c1) {
    int idx = blockIdx.x * 128 + threadIdx.x;
    int bh = idx >> 9, i = idx & 511;
    int b = bh >> 4, h = bh & 15;
    float a[5];
#pragma unroll
    for (int p = 0; p < 5; ++p) a[p] = atc1[((size_t)b * LSEQ + i) * 5 + p];
#pragma unroll
    for (int q = 0; q < 5; ++q) {
        float s = 0.f;
#pragma unroll
        for (int p = 0; p < 5; ++p) s += a[p] * U[(h * 5 + p) * 5 + q];
        c1[(size_t)idx * 5 + q] = s;
    }
}

// ---------------------------------------------------------------------------
// QKV projection. grid (16,64,3), block 128 (4 waves). A direct from global;
// B double-buffered in LDS via async DMA overlapped with WMMA.
// ---------------------------------------------------------------------------
__global__ __launch_bounds__(128) void gemm_qkv(
    const unsigned short* __restrict__ xb1, const unsigned short* __restrict__ xb2,
    const unsigned short* __restrict__ wq, const unsigned short* __restrict__ wk,
    const unsigned short* __restrict__ wv,
    const float* __restrict__ bq, const float* __restrict__ bk,
    const float* __restrict__ bv,
    unsigned short* __restrict__ qb, unsigned short* __restrict__ kb,
    unsigned short* __restrict__ vt) {

    const int z = blockIdx.z;
    const unsigned short* X = (z == 0) ? xb1 : xb2;
    const unsigned short* W = (z == 0) ? wq : (z == 1 ? wk : wv);
    const float* bias = (z == 0) ? bq : (z == 1 ? bk : bv);

    __shared__ __align__(16) unsigned short wb_s[2][64 * STR32];

    const int tid = threadIdx.x, lane = tid & 31, wz = tid >> 5;
    const int lm = lane & 15, hi = lane >> 4;
    const int rowM0 = blockIdx.y * 64, colN0 = blockIdx.x * 64;

    const int sr = tid >> 1, sq = (tid & 1) * 16;
    const unsigned ld0 = lds_off(&wb_s[0][sr * STR32 + sq]);
    const unsigned ld1 = lds_off(&wb_s[1][sr * STR32 + sq]);
    const unsigned gwrow = (unsigned)((colN0 + sr) * DIMC + sq) * 2u;

    const unsigned short* Arow = X + (size_t)(rowM0 + wz * 16 + lm) * DIMC;

    f32x8 zro = {0.f, 0.f, 0.f, 0.f, 0.f, 0.f, 0.f, 0.f};
    f32x8 acc[4];
#pragma unroll
    for (int t = 0; t < 4; ++t) acc[t] = zro;

    async_g2l_b128(ld0,      W, gwrow);
    async_g2l_b128(ld0 + 16, W, gwrow + 16);

    for (int kc = 0; kc < DIMC; kc += 32) {
        const int cur = (kc >> 5) & 1;
        wait_async0();
        __syncthreads();
        if (kc + 32 < DIMC) {
            unsigned l = cur ? ld0 : ld1;
            unsigned g = gwrow + (unsigned)(kc + 32) * 2u;
            async_g2l_b128(l,      W, g);
            async_g2l_b128(l + 16, W, g + 16);
        }
        // load ALL fragments first, then back-to-back WMMAs
        FragU a;
        a.h[0] = *(const bf16x8*)(Arow + kc + hi * 16);
        a.h[1] = *(const bf16x8*)(Arow + kc + hi * 16 + 8);
        bf16x16 bfr[4];
#pragma unroll
        for (int t = 0; t < 4; ++t)
            bfr[t] = *(const bf16x16*)&wb_s[cur][(t * 16 + lm) * STR32 + hi * 16];
#pragma unroll
        for (int t = 0; t < 4; ++t)
            acc[t] = wmma_bf16(a.f, bfr[t], acc[t]);
    }

#pragma unroll
    for (int t = 0; t < 4; ++t) {
        int ng = colN0 + t * 16 + lm;
        float bv_ = bias[ng];
        int hh = ng >> 6, d = ng & 63;
#pragma unroll
        for (int e = 0; e < 8; ++e) {
            int rg = rowM0 + wz * 16 + e + 8 * hi;
            int bb = rg >> 9, ii = rg & 511;
            unsigned short v16 = f2bf(acc[t][e] + bv_);
            if (z == 0)      qb[(((size_t)bb * HN + hh) * LSEQ + ii) * HD + d] = v16;
            else if (z == 1) kb[(((size_t)bb * HN + hh) * LSEQ + ii) * HD + d] = v16;
            else             vt[(((size_t)bb * HN + hh) * HD + d) * LSEQ + ii] = v16;
        }
    }
}

// ---------------------------------------------------------------------------
// Streaming dual-softmax attention. grid (8,128), block 128 (4 waves).
// K chunks via TDM tensor DMA (padded-LDS descriptor); V^T via async DMA;
// both double-buffered and overlapped with WMMA.
// ---------------------------------------------------------------------------
__global__ __launch_bounds__(128) void attn_kernel(
    const unsigned short* __restrict__ qb, const unsigned short* __restrict__ kb,
    const unsigned short* __restrict__ vt, const float* __restrict__ c1,
    const float* __restrict__ atc2, const float* __restrict__ mixp,
    unsigned short* __restrict__ ob) {

    __shared__ __align__(16) unsigned short kb_s[2][32 * STR64];
    __shared__ __align__(16) unsigned short vt_s[2][64 * STR32];
    __shared__ float a2_s[2][32][6];
    __shared__ __align__(16) unsigned short ps_s[4][16 * STR32];
    __shared__ __align__(16) unsigned short pb_s[4][16 * STR32];

    const int tid = threadIdx.x, lane = tid & 31, w = tid >> 5;
    const int lm = lane & 15, hi = lane >> 4;
    const int bh = blockIdx.y, b = bh >> 4, h = bh & 15;
    const int i0 = blockIdx.x * 64;

    FragU qa[2];
    {
        const unsigned short* qrow = qb + ((size_t)bh * LSEQ + i0 + w * 16 + lm) * HD;
#pragma unroll
        for (int f = 0; f < 2; ++f) {
            qa[f].h[0] = *(const bf16x8*)(qrow + f * 32 + hi * 16);
            qa[f].h[1] = *(const bf16x8*)(qrow + f * 32 + hi * 16 + 8);
        }
    }
    float c1r[8][5];
#pragma unroll
    for (int e = 0; e < 8; ++e) {
        const float* cp = c1 + ((size_t)bh * LSEQ + i0 + w * 16 + e + 8 * hi) * 5;
#pragma unroll
        for (int p = 0; p < 5; ++p) c1r[e][p] = cp[p];
    }

    f32x8 zro = {0.f, 0.f, 0.f, 0.f, 0.f, 0.f, 0.f, 0.f};
    f32x8 Os[4], Obio[4];
#pragma unroll
    for (int t = 0; t < 4; ++t) { Os[t] = zro; Obio[t] = zro; }
    float ls[8], lb[8];
#pragma unroll
    for (int e = 0; e < 8; ++e) { ls[e] = 0.f; lb[e] = 0.f; }

    const unsigned short* kbase = kb + (size_t)bh * LSEQ * HD;
#if !HAS_TDM
    const int kr = tid >> 2, kq = (tid & 3) * 8;
    const unsigned kld[2] = { lds_off(&kb_s[0][kr * STR64 + kq]),
                              lds_off(&kb_s[1][kr * STR64 + kq]) };
    const unsigned kgrow = (unsigned)((kr * HD + kq) * 2);
#endif
    const int vr = tid >> 1, vq = (tid & 1) * 16;
    const unsigned vld[2] = { lds_off(&vt_s[0][vr * STR32 + vq]),
                              lds_off(&vt_s[1][vr * STR32 + vq]) };
    const unsigned vgrow = (unsigned)(((bh * HD + vr) * LSEQ + vq) * 2);

    // prologue: DMA chunk 0, stage atc2 chunk 0
#if HAS_TDM
    if (tid < 32) tdm_load_k(lds_off(&kb_s[0][0]), kbase);
#else
    async_g2l_b128(kld[0],       kbase, kgrow);
    async_g2l_b128(kld[0] + 128, kbase, kgrow + 128);
#endif
    async_g2l_b128(vld[0],      vt, vgrow);
    async_g2l_b128(vld[0] + 16, vt, vgrow + 16);
    for (int idx = tid; idx < 160; idx += 128)
        a2_s[0][idx / 5][idx % 5] = atc2[((size_t)b * LSEQ) * 5 + idx];

    for (int jc = 0; jc < LSEQ; jc += 32) {
        const int cur = (jc >> 5) & 1, nxt = cur ^ 1;
        wait_async0();
        wait_tensor0();
        __syncthreads();
        if (jc + 32 < LSEQ) {
#if HAS_TDM
            if (tid < 32)
                tdm_load_k(lds_off(&kb_s[nxt][0]), kbase + (size_t)(jc + 32) * HD);
#else
            unsigned gofs = (unsigned)((jc + 32) * HD * 2);
            async_g2l_b128(kld[nxt],       kbase, kgrow + gofs);
            async_g2l_b128(kld[nxt] + 128, kbase, kgrow + gofs + 128);
#endif
            unsigned vofs = (unsigned)((jc + 32) * 2);
            async_g2l_b128(vld[nxt],      vt, vgrow + vofs);
            async_g2l_b128(vld[nxt] + 16, vt, vgrow + vofs + 16);
            for (int idx = tid; idx < 160; idx += 128)
                a2_s[nxt][idx / 5][idx % 5] =
                    atc2[((size_t)b * LSEQ + jc + 32) * 5 + idx];
        }

        // scores (two 16-col tiles), exp, stash P tiles in per-wave LDS
#pragma unroll
        for (int jt = 0; jt < 2; ++jt) {
            int jrow = jt * 16 + lm;
            bf16x16 kf0 = *(const bf16x16*)&kb_s[cur][jrow * STR64 + hi * 16];
            bf16x16 kf1 = *(const bf16x16*)&kb_s[cur][jrow * STR64 + 32 + hi * 16];
            f32x8 S = zro;
            S = wmma_bf16(qa[0].f, kf0, S);
            S = wmma_bf16(qa[1].f, kf1, S);
#pragma unroll
            for (int e = 0; e < 8; ++e) {
                float pstd = __expf(S[e] * 0.125f);     // 1/sqrt(64)
                ls[e] += pstd;
                ps_s[w][(e + 8 * hi) * STR32 + jrow] = f2bf(pstd);
                float sb = c1r[e][0] * a2_s[cur][jrow][0];
                sb += c1r[e][1] * a2_s[cur][jrow][1];
                sb += c1r[e][2] * a2_s[cur][jrow][2];
                sb += c1r[e][3] * a2_s[cur][jrow][3];
                sb += c1r[e][4] * a2_s[cur][jrow][4];
                float pbio = __expf(sb);
                lb[e] += pbio;
                pb_s[w][(e + 8 * hi) * STR32 + jrow] = f2bf(pbio);
            }
        }

        // P @ V for both branches: load all fragments, then 8 WMMAs
        FragU pa, pbf;
        pa.h[0]  = *(const bf16x8*)&ps_s[w][lm * STR32 + hi * 16];
        pa.h[1]  = *(const bf16x8*)&ps_s[w][lm * STR32 + hi * 16 + 8];
        pbf.h[0] = *(const bf16x8*)&pb_s[w][lm * STR32 + hi * 16];
        pbf.h[1] = *(const bf16x8*)&pb_s[w][lm * STR32 + hi * 16 + 8];
        bf16x16 vf[4];
#pragma unroll
        for (int t = 0; t < 4; ++t)
            vf[t] = *(const bf16x16*)&vt_s[cur][(t * 16 + lm) * STR32 + hi * 16];
#pragma unroll
        for (int t = 0; t < 4; ++t) {
            Os[t]   = wmma_bf16(pa.f,  vf[t], Os[t]);
            Obio[t] = wmma_bf16(pbf.f, vf[t], Obio[t]);
        }
    }

    // row sums: reduce across the 16 lanes of each C-layout half (wave32)
#pragma unroll
    for (int e = 0; e < 8; ++e) {
#pragma unroll
        for (int off = 1; off < 16; off <<= 1) {
            ls[e] += __shfl_xor(ls[e], off);
            lb[e] += __shfl_xor(lb[e], off);
        }
    }
    float mix = tanhf(mixp[0]);
    float rr  = (mix + 1.f) * 0.5f;
#pragma unroll
    for (int t = 0; t < 4; ++t) {
        int col = h * HD + t * 16 + lm;
#pragma unroll
        for (int e = 0; e < 8; ++e) {
            int irow = i0 + w * 16 + e + 8 * hi;
            float val = (1.f - rr) * (Os[t][e] / ls[e]) + rr * (Obio[t][e] / lb[e]);
            ob[((size_t)b * LSEQ + irow) * DIMC + col] = f2bf(val);
        }
    }
}

// ---------------------------------------------------------------------------
// Output projection: out f32 = ob(bf16) @ Wo^T(bf16) + bo. grid (16,64).
// ---------------------------------------------------------------------------
__global__ __launch_bounds__(128) void gemm_out(
    const unsigned short* __restrict__ A, const unsigned short* __restrict__ W,
    const float* __restrict__ bias, float* __restrict__ out) {

    __shared__ __align__(16) unsigned short wb_s[2][64 * STR32];

    const int tid = threadIdx.x, lane = tid & 31, wz = tid >> 5;
    const int lm = lane & 15, hi = lane >> 4;
    const int rowM0 = blockIdx.y * 64, colN0 = blockIdx.x * 64;

    const int sr = tid >> 1, sq = (tid & 1) * 16;
    const unsigned ld0 = lds_off(&wb_s[0][sr * STR32 + sq]);
    const unsigned ld1 = lds_off(&wb_s[1][sr * STR32 + sq]);
    const unsigned gwrow = (unsigned)((colN0 + sr) * DIMC + sq) * 2u;

    const unsigned short* Arow = A + (size_t)(rowM0 + wz * 16 + lm) * DIMC;

    f32x8 zro = {0.f, 0.f, 0.f, 0.f, 0.f, 0.f, 0.f, 0.f};
    f32x8 acc[4];
#pragma unroll
    for (int t = 0; t < 4; ++t) acc[t] = zro;

    async_g2l_b128(ld0,      W, gwrow);
    async_g2l_b128(ld0 + 16, W, gwrow + 16);

    for (int kc = 0; kc < DIMC; kc += 32) {
        const int cur = (kc >> 5) & 1;
        wait_async0();
        __syncthreads();
        if (kc + 32 < DIMC) {
            unsigned l = cur ? ld0 : ld1;
            unsigned g = gwrow + (unsigned)(kc + 32) * 2u;
            async_g2l_b128(l,      W, g);
            async_g2l_b128(l + 16, W, g + 16);
        }
        FragU a;
        a.h[0] = *(const bf16x8*)(Arow + kc + hi * 16);
        a.h[1] = *(const bf16x8*)(Arow + kc + hi * 16 + 8);
        bf16x16 bfr[4];
#pragma unroll
        for (int t = 0; t < 4; ++t)
            bfr[t] = *(const bf16x16*)&wb_s[cur][(t * 16 + lm) * STR32 + hi * 16];
#pragma unroll
        for (int t = 0; t < 4; ++t)
            acc[t] = wmma_bf16(a.f, bfr[t], acc[t]);
    }

#pragma unroll
    for (int t = 0; t < 4; ++t) {
        int ng = colN0 + t * 16 + lm;
        float bv_ = bias[ng];
#pragma unroll
        for (int e = 0; e < 8; ++e) {
            int rg = rowM0 + wz * 16 + e + 8 * hi;
            out[(size_t)rg * DIMC + ng] = acc[t][e] + bv_;
        }
    }
}

// ---------------------------------------------------------------------------
extern "C" void kernel_launch(void* const* d_in, const int* in_sizes, int n_in,
                              void* d_out, int out_size, void* d_ws, size_t ws_size,
                              hipStream_t stream) {
    const float* seq1 = (const float*)d_in[0];
    const float* seq2 = (const float*)d_in[1];
    const float* atc1 = (const float*)d_in[2];
    const float* atc2 = (const float*)d_in[3];
    const float* Wq   = (const float*)d_in[4];
    const float* bq   = (const float*)d_in[5];
    const float* Wk   = (const float*)d_in[6];
    const float* bk   = (const float*)d_in[7];
    const float* Wv   = (const float*)d_in[8];
    const float* bv   = (const float*)d_in[9];
    const float* Wo   = (const float*)d_in[10];
    const float* bo   = (const float*)d_in[11];
    const float* U    = (const float*)d_in[12];
    const float* mixp = (const float*)d_in[13];
    float* out = (float*)d_out;

    char* ws = (char*)d_ws;
    unsigned short* qb  = (unsigned short*)(ws + QB_OFF);
    unsigned short* kb  = (unsigned short*)(ws + KB_OFF);
    unsigned short* vt  = (unsigned short*)(ws + VT_OFF);
    unsigned short* ob  = (unsigned short*)(ws + OB_OFF);
    float*          c1  = (float*)(ws + C1_OFF);
    unsigned short* xb1 = (unsigned short*)(ws + XB1_OFF);
    unsigned short* xb2 = (unsigned short*)(ws + XB2_OFF);
    unsigned short* wqb = (unsigned short*)(ws + WQ_OFF);
    unsigned short* wkb = (unsigned short*)(ws + WK_OFF);
    unsigned short* wvb = (unsigned short*)(ws + WV_OFF);
    unsigned short* wob = (unsigned short*)(ws + WO_OFF);

    cvt_kernel<<<dim3(2048), 256, 0, stream>>>(seq1, xb1);
    cvt_kernel<<<dim3(2048), 256, 0, stream>>>(seq2, xb2);
    cvt_kernel<<<dim3(512),  256, 0, stream>>>(Wq, wqb);
    cvt_kernel<<<dim3(512),  256, 0, stream>>>(Wk, wkb);
    cvt_kernel<<<dim3(512),  256, 0, stream>>>(Wv, wvb);
    cvt_kernel<<<dim3(512),  256, 0, stream>>>(Wo, wob);

    c1_kernel<<<dim3(512), 128, 0, stream>>>(atc1, U, c1);
    gemm_qkv<<<dim3(16, 64, 3), 128, 0, stream>>>(xb1, xb2, wqb, wkb, wvb,
                                                  bq, bk, bv, qb, kb, vt);
    attn_kernel<<<dim3(8, 128), 128, 0, stream>>>(qb, kb, vt, c1, atc2, mixp, ob);
    gemm_out<<<dim3(16, 64), 128, 0, stream>>>(ob, wob, bo, out);
}
